// Intra_Modal_Modeling_Lite_20495583936767
// MI455X (gfx1250) — compile-verified, matrix operand order checked by reading, not run on hardware
//
#include <hip/hip_runtime.h>
#include <cmath>

// ---------------------------------------------------------------------------
// Types for CDNA5 WMMA (wave32, V_WMMA_F32_16X16X32_BF16)
// ---------------------------------------------------------------------------
typedef __attribute__((ext_vector_type(8)))  __bf16 bf16x8;
typedef __attribute__((ext_vector_type(16))) __bf16 bf16x16;
typedef __attribute__((ext_vector_type(8)))  float  f32x8;

#define NEGBIG -1.0e9f

__device__ __forceinline__ float gelu_exact(float v) {
    return 0.5f * v * (1.0f + erff(v * 0.70710678118654752f));
}

// ---------------------------------------------------------------------------
// Batched GEMM:  C[z] = epilogue(alpha * A[z] @ op(B[z]) + bias) (+ residual)
//   TRANS_B=true : B stored [N,K] row-major (x @ W^T, the torch Linear case)
//   TRANS_B=false: B stored [K,N] row-major (probs @ V case)
//   z decomposes as (zb, zh) = (z/H, z%H); per-matrix offset zb*s1 + zh*s2.
//   EP: 0 = none, 1 = exact GELU, 2 = sim-clip ((v+1)/2 clamped to [0,1])
// Block: 128 threads = 4 waves in a 2x2 grid; each wave does 2x2 WMMA tiles.
// Block tile 64x64, K-step 32. Software pipeline: global float4 loads for
// tile k+1 are in flight while WMMAs consume LDS tile k (ping-pong buffers,
// one barrier per K-step).
// Requires M%64==0, N%64==0, K%32==0 and 16B-aligned A/B tiles (true here).
// ---------------------------------------------------------------------------
template <bool TRANS_B, int EP>
__global__ __launch_bounds__(128)
void wmma_gemm(const float* __restrict__ A, int lda, long long sA1, long long sA2,
               const float* __restrict__ Bm, int ldb, long long sB1, long long sB2,
               float* __restrict__ C, int ldc, long long sC1, long long sC2,
               const float* __restrict__ bias, const float* __restrict__ res,
               int M, int N, int K, float alpha, int H)
{
    __shared__ __align__(16) __bf16 As[2][64 * 48];
    __shared__ __align__(16) __bf16 Bs[2][64 * 48];

    const int tid = threadIdx.x;
    const int z  = blockIdx.z;
    const int zb = z / H, zh = z % H;
    const float* Ap = A  + zb * sA1 + zh * sA2;
    const float* Bp = Bm + zb * sB1 + zh * sB2;
    float*       Cp = C  + zb * sC1 + zh * sC2;
    const float* Rp = res ? (res + zb * sC1 + zh * sC2) : nullptr;

    const int m0 = blockIdx.y * 64;
    const int n0 = blockIdx.x * 64;

    // ---- load/store helper indices ----
    const int lrow = tid >> 1;            // 0..63
    const int lcol = (tid & 1) * 16;      // 0 or 16
    const int bkk  = tid >> 2;            // 0..31 (NN path)
    const int bnb  = (tid & 3) * 16;      // 0,16,32,48 (NN path)

    float4 ar[4], br[4];

    auto load_tiles = [&](int k0) {
        const float4* sa =
            reinterpret_cast<const float4*>(Ap + (long long)(m0 + lrow) * lda + k0 + lcol);
#pragma unroll
        for (int j = 0; j < 4; ++j) ar[j] = sa[j];
        if (TRANS_B) {
            const float4* sb =
                reinterpret_cast<const float4*>(Bp + (long long)(n0 + lrow) * ldb + k0 + lcol);
#pragma unroll
            for (int j = 0; j < 4; ++j) br[j] = sb[j];
        } else {
            const float4* sb =
                reinterpret_cast<const float4*>(Bp + (long long)(k0 + bkk) * ldb + n0 + bnb);
#pragma unroll
            for (int j = 0; j < 4; ++j) br[j] = sb[j];
        }
    };

    auto store_tiles = [&](int buf) {
        { // A: two 16-byte bf16 stores
            __bf16* dst = &As[buf][lrow * 48 + lcol];
#pragma unroll
            for (int c = 0; c < 2; ++c) {
                bf16x8 v;
                const float* f = &ar[c * 2].x;
#pragma unroll
                for (int j = 0; j < 8; ++j) v[j] = (__bf16)f[j];
                *(bf16x8*)(dst + c * 8) = v;
            }
        }
        if (TRANS_B) {
            __bf16* dst = &Bs[buf][lrow * 48 + lcol];
#pragma unroll
            for (int c = 0; c < 2; ++c) {
                bf16x8 v;
                const float* f = &br[c * 2].x;
#pragma unroll
                for (int j = 0; j < 8; ++j) v[j] = (__bf16)f[j];
                *(bf16x8*)(dst + c * 8) = v;
            }
        } else { // transpose on store: Bs[n][k]
            const float* f = &br[0].x;
#pragma unroll
            for (int j = 0; j < 16; ++j) Bs[buf][(bnb + j) * 48 + bkk] = (__bf16)f[j];
        }
    };

    // ---- accumulators / compute indices ----
    f32x8 acc[2][2];
#pragma unroll
    for (int i = 0; i < 2; ++i)
#pragma unroll
        for (int j = 0; j < 2; ++j) acc[i][j] = (f32x8)0.0f;

    const int wave = tid >> 5;
    const int lane = tid & 31;
    const int wr = (wave >> 1) * 32;   // wave row origin in block tile
    const int wc = (wave & 1)  * 32;   // wave col origin in block tile
    const int rrow = lane & 15;        // M (or N) index within 16x16 tile
    const int cb   = (lane >> 4) * 8;  // K chunk base (0 or 8)

    const int nsteps = K >> 5;

    // prologue: tile 0 -> LDS buf 0
    load_tiles(0);
    store_tiles(0);
    __syncthreads();

    for (int step = 0; step < nsteps; ++step) {
        const int cur = step & 1;
        const bool more = (step + 1) < nsteps;
        if (more) load_tiles((step + 1) << 5);   // global loads in flight over WMMAs

        // Per-lane operands per ISA 16-bit layout:
        // lane<16 holds K {0..7,16..23}; lane>=16 holds K {8..15,24..31}.
        bf16x16 av[2], bv[2];
#pragma unroll
        for (int mt = 0; mt < 2; ++mt) {
            const __bf16* p = &As[cur][(wr + mt * 16 + rrow) * 48 + cb];
            bf16x8 lo = *(const bf16x8*)p;
            bf16x8 hi = *(const bf16x8*)(p + 16);
#pragma unroll
            for (int j = 0; j < 8; ++j) { av[mt][j] = lo[j]; av[mt][j + 8] = hi[j]; }
        }
#pragma unroll
        for (int nt = 0; nt < 2; ++nt) {
            const __bf16* p = &Bs[cur][(wc + nt * 16 + rrow) * 48 + cb];
            bf16x8 lo = *(const bf16x8*)p;
            bf16x8 hi = *(const bf16x8*)(p + 16);
#pragma unroll
            for (int j = 0; j < 8; ++j) { bv[nt][j] = lo[j]; bv[nt][j + 8] = hi[j]; }
        }
#pragma unroll
        for (int mt = 0; mt < 2; ++mt)
#pragma unroll
            for (int nt = 0; nt < 2; ++nt)
                acc[mt][nt] = __builtin_amdgcn_wmma_f32_16x16x32_bf16(
                    false, av[mt], false, bv[nt], (short)0, acc[mt][nt], false, false);

        if (more) store_tiles(cur ^ 1);          // fill the other buffer
        __syncthreads();
    }

    // Epilogue. C/D layout: lane&15 = N, (lane>>4)*8 + vgpr = M.
    const int nl = lane & 15;
    const int mb = (lane >> 4) * 8;
#pragma unroll
    for (int mt = 0; mt < 2; ++mt) {
#pragma unroll
        for (int nt = 0; nt < 2; ++nt) {
            int gn = n0 + wc + nt * 16 + nl;
            float bval = bias ? bias[gn] : 0.0f;
#pragma unroll
            for (int v = 0; v < 8; ++v) {
                int gm = m0 + wr + mt * 16 + mb + v;
                float val = acc[mt][nt][v] * alpha + bval;
                if (EP == 1) val = gelu_exact(val);
                if (EP == 2) { val = (val + 1.0f) * 0.5f; val = fminf(fmaxf(val, 0.0f), 1.0f); }
                long long ci = (long long)gm * ldc + gn;
                if (Rp) val += Rp[ci];
                Cp[ci] = val;
            }
        }
    }
}

// ---------------------------------------------------------------------------
// LayerNorm over last dim E (one block per row), optional fused exact GELU.
// ---------------------------------------------------------------------------
__global__ __launch_bounds__(128)
void ln_kernel(const float* __restrict__ x, float* __restrict__ y,
               const float* __restrict__ g, const float* __restrict__ b,
               int E, int gelu_out)
{
    __shared__ float red[128];
    long long row = blockIdx.x;
    const float* xr = x + row * E;
    float* yr = y + row * E;
    int tid = threadIdx.x;
    float s = 0.f, sq = 0.f;
    for (int i = tid; i < E; i += 128) { float v = xr[i]; s += v; sq += v * v; }
    red[tid] = s; __syncthreads();
    for (int st = 64; st > 0; st >>= 1) { if (tid < st) red[tid] += red[tid + st]; __syncthreads(); }
    float mean = red[0] / (float)E; __syncthreads();
    red[tid] = sq; __syncthreads();
    for (int st = 64; st > 0; st >>= 1) { if (tid < st) red[tid] += red[tid + st]; __syncthreads(); }
    float var = red[0] / (float)E - mean * mean;
    float rstd = rsqrtf(var + 1e-5f);
    for (int i = tid; i < E; i += 128) {
        float v = (xr[i] - mean) * rstd * g[i] + b[i];
        if (gelu_out) v = gelu_exact(v);
        yr[i] = v;
    }
}

// y = x + pos (pos broadcast with period `per`)
__global__ void addpos_kernel(const float* __restrict__ x, const float* __restrict__ pos,
                              float* __restrict__ y, long long total, long long per)
{
    long long i = (long long)blockIdx.x * blockDim.x + threadIdx.x;
    if (i < total) y[i] = x[i] + pos[i % per];
}

// In-place masked softmax over rows of length 512 (N fixed).
// keypad: bool[B,N] -> add NEGBIG where true; madd: float[B,N,N] additive.
__global__ __launch_bounds__(256)
void softmax_kernel(float* __restrict__ sc, const unsigned char* __restrict__ keypad,
                    const float* __restrict__ madd, int N, int H)
{
    __shared__ float red[256];
    int q = blockIdx.x, z = blockIdx.y;
    int b = z / H;
    float* row = sc + ((long long)z * N + q) * N;
    int tid = threadIdx.x;
    float v[2];
#pragma unroll
    for (int j = 0; j < 2; ++j) {
        int k = tid + j * 256;
        float t = row[k];
        if (keypad) t += keypad[(long long)b * N + k] ? NEGBIG : 0.0f;
        if (madd)   t += madd[((long long)b * N + q) * N + k];
        v[j] = t;
    }
    float m = fmaxf(v[0], v[1]);
    red[tid] = m; __syncthreads();
    for (int st = 128; st > 0; st >>= 1) { if (tid < st) red[tid] = fmaxf(red[tid], red[tid + st]); __syncthreads(); }
    m = red[0]; __syncthreads();
    float e0 = __expf(v[0] - m), e1 = __expf(v[1] - m);
    red[tid] = e0 + e1; __syncthreads();
    for (int st = 128; st > 0; st >>= 1) { if (tid < st) red[tid] += red[tid + st]; __syncthreads(); }
    float inv = 1.0f / red[0];
    row[tid] = e0 * inv;
    row[tid + 256] = e1 * inv;
}

// Row-wise L2 normalize, D=64 (one thread per element).
__global__ __launch_bounds__(64)
void l2norm_kernel(const float* __restrict__ cf, float* __restrict__ nv, int D)
{
    __shared__ float red[64];
    long long row = blockIdx.x;
    int tid = threadIdx.x;
    float v = cf[row * D + tid];
    red[tid] = v * v; __syncthreads();
    for (int st = 32; st > 0; st >>= 1) { if (tid < st) red[tid] += red[tid + st]; __syncthreads(); }
    nv[row * D + tid] = v * rsqrtf(red[0]);
}

// flag = (sum(mask) > 0)
__global__ __launch_bounds__(256)
void cond_kernel(const unsigned char* __restrict__ mask, int n, int* __restrict__ flag)
{
    __shared__ int red[256];
    int tid = threadIdx.x, s = 0;
    for (int i = tid; i < n; i += 256) s += mask[i] ? 1 : 0;
    red[tid] = s; __syncthreads();
    for (int st = 128; st > 0; st >>= 1) { if (tid < st) red[tid] += red[tid + st]; __syncthreads(); }
    if (tid == 0) *flag = red[0] > 0 ? 1 : 0;
}

// Top-Ksel selection per (b,q) row of sim -> additive attention mask.
// minimize=0: top-K largest of (off?-1:sim) - eye; minimize=1: K smallest of (off?2:sim).
// Tie-break matches lax.top_k: larger value first, then smaller index.
__global__ __launch_bounds__(128)
void topk_mask_kernel(const float* __restrict__ sim, const unsigned char* __restrict__ mm,
                      const int* __restrict__ condp, float* __restrict__ madd,
                      int N, int Ksel, int minimize)
{
    __shared__ float vals[512];
    __shared__ unsigned char allowed[512];
    __shared__ float rv[128];
    __shared__ int   ri[128];
    int q = blockIdx.x, b = blockIdx.y, tid = threadIdx.x;
    int cond = *condp;
    long long base = ((long long)b * N + q) * N;
    for (int k = tid; k < N; k += 128) {
        int off = cond && (mm[base + k] == 0);
        float s = sim[base + k];
        float cand;
        if (minimize) { cand = off ? 2.0f : s; cand = -cand; }
        else          { cand = off ? -1.0f : s; if (k == q) cand -= 1.0f; }
        vals[k] = cand;
        allowed[k] = 0;
    }
    __syncthreads();
    for (int t = 0; t < Ksel; ++t) {
        float bv = -3.0e38f; int bi = N;
        for (int k = tid; k < N; k += 128) { float v = vals[k]; if (v > bv) { bv = v; bi = k; } }
        rv[tid] = bv; ri[tid] = bi; __syncthreads();
        for (int st = 64; st > 0; st >>= 1) {
            if (tid < st) {
                float ov = rv[tid + st]; int oi = ri[tid + st];
                if (ov > rv[tid] || (ov == rv[tid] && oi < ri[tid])) { rv[tid] = ov; ri[tid] = oi; }
            }
            __syncthreads();
        }
        if (tid == 0) { allowed[ri[0]] = 1; vals[ri[0]] = -3.0e38f; }
        __syncthreads();
    }
    for (int k = tid; k < N; k += 128)
        madd[base + k] = allowed[k] ? 0.0f : NEGBIG;
}

// ---------------------------------------------------------------------------
// Host-side launch helpers
// ---------------------------------------------------------------------------
template <bool TB, int EP>
static void gemm(hipStream_t st, const float* A, int lda, long long sA1, long long sA2,
                 const float* Bm, int ldb, long long sB1, long long sB2,
                 float* C, int ldc, long long sC1, long long sC2,
                 const float* bias, const float* res,
                 int M, int Nn, int K, float alpha, int Z, int Hh)
{
    dim3 g(Nn / 64, M / 64, Z);
    wmma_gemm<TB, EP><<<g, 128, 0, st>>>(A, lda, sA1, sA2, Bm, ldb, sB1, sB2,
                                         C, ldc, sC1, sC2, bias, res, M, Nn, K, alpha, Hh);
}

extern "C" void kernel_launch(void* const* d_in, const int* in_sizes, int n_in,
                              void* d_out_, int out_size, void* d_ws, size_t ws_size,
                              hipStream_t s)
{
    const int Bn = 16, Nn = 512, E = 512, HID = 2048, HSI = 8, HAG = 4, TOK = 64;
    const int MT = Bn * Nn; // 8192 tokens

    auto F = [&](int i) { return (const float*)d_in[i]; };
    const float* feats = F(0);
    const unsigned char* keymask = (const unsigned char*)d_in[1]; // mask[B,N], True=padded
    const unsigned char* mmask   = (const unsigned char*)d_in[2]; // matrix_mask[B,N,N]
    const float* pos = F(3);

    float* out = (float*)d_out_;
    float* out_feats = out;                          // [B,N,E]
    float* out_sim   = out + (long long)MT * E;      // [B,N,N]
    float* out_cf    = out + 2LL * MT * E;           // [B,N,64]

    // workspace layout
    float* w = (float*)d_ws;
    float* x    = w; w += (long long)MT * E;
    float* y    = w; w += (long long)MT * E;
    float* t    = w; w += (long long)MT * E;
    float* qkvb = w; w += (long long)MT * 3 * E;
    float* big  = w; w += (long long)Bn * HSI * Nn * Nn; // scores (also FFN hidden)
    float* madd = w; w += (long long)Bn * Nn * Nn;
    float* cf1  = w; w += (long long)MT * 128;
    float* cf2  = w; w += (long long)MT * 128;
    float* nv   = w; w += (long long)MT * 64;
    int*   flag = (int*)w;

    // MHA: q_in==k_in (= qkin), v from vin; writes outC = o@Wo^T + bo (+ resid)
    auto mha = [&](const float* qkin, const float* vin,
                   const float* Wqkv, const float* bqkv,
                   const float* Wo, const float* bo, int Hh,
                   const unsigned char* kp, const float* ma,
                   const float* resid, float* outC)
    {
        int d = E / Hh;
        if (qkin == vin) {
            gemm<true, 0>(s, qkin, E, 0, 0, Wqkv, E, 0, 0, qkvb, 3 * E, 0, 0,
                          bqkv, nullptr, MT, 3 * E, E, 1.f, 1, 1);
        } else {
            gemm<true, 0>(s, qkin, E, 0, 0, Wqkv, E, 0, 0, qkvb, 3 * E, 0, 0,
                          bqkv, nullptr, MT, 2 * E, E, 1.f, 1, 1);
            gemm<true, 0>(s, vin, E, 0, 0, Wqkv + (long long)2 * E * E, E, 0, 0,
                          qkvb + 2 * E, 3 * E, 0, 0, bqkv + 2 * E, nullptr,
                          MT, E, E, 1.f, 1, 1);
        }
        long long sTok = (long long)Nn * 3 * E;
        // scores = alpha * Q @ K^T  (batched over B*H)
        gemm<true, 0>(s, qkvb, 3 * E, sTok, d, qkvb + E, 3 * E, sTok, d,
                      big, Nn, (long long)Hh * Nn * Nn, (long long)Nn * Nn,
                      nullptr, nullptr, Nn, Nn, d, 1.0f / sqrtf((float)d), Bn * Hh, Hh);
        softmax_kernel<<<dim3(Nn, Bn * Hh), 256, 0, s>>>(big, kp, ma, Nn, Hh);
        // o_heads = probs @ V  -> t arranged [b*N + q, h*d + dd]
        gemm<false, 0>(s, big, Nn, (long long)Hh * Nn * Nn, (long long)Nn * Nn,
                       qkvb + 2 * E, 3 * E, sTok, d,
                       t, E, (long long)Nn * E, d,
                       nullptr, nullptr, Nn, d, Nn, 1.f, Bn * Hh, Hh);
        gemm<true, 0>(s, t, E, 0, 0, Wo, E, 0, 0, outC, E, 0, 0,
                      bo, resid, MT, E, E, 1.f, 1, 1);
    };

    // ---- input projection ----
    gemm<true, 0>(s, feats, 512, 0, 0, F(4), 512, 0, 0, x, E, 0, 0,
                  F(5), nullptr, MT, E, 512, 1.f, 1, 1);

    // ---- 2 Self_Interaction blocks ----
    const int lbase[2] = {6, 18};
    for (int L = 0; L < 2; ++L) {
        int p = lbase[L];
        ln_kernel<<<MT, 128, 0, s>>>(x, y, F(p + 0), F(p + 1), E, 0);
        addpos_kernel<<<(MT * E + 255) / 256, 256, 0, s>>>(y, pos, t, (long long)MT * E, (long long)Nn * E);
        mha(t, y, F(p + 2), F(p + 3), F(p + 4), F(p + 5), HSI, keymask, nullptr, x, x);
        ln_kernel<<<MT, 128, 0, s>>>(x, y, F(p + 6), F(p + 7), E, 0);
        gemm<true, 1>(s, y, E, 0, 0, F(p + 8), E, 0, 0, big, HID, 0, 0,
                      F(p + 9), nullptr, MT, HID, E, 1.f, 1, 1);            // GELU fused
        gemm<true, 0>(s, big, HID, 0, 0, F(p + 10), HID, 0, 0, x, E, 0, 0,
                      F(p + 11), x, MT, E, HID, 1.f, 1, 1);                 // + residual
    }

    // ---- consist encoder + cosine similarity ----
    gemm<true, 0>(s, x, E, 0, 0, F(30), E, 0, 0, cf1, 128, 0, 0, F(31), nullptr, MT, 128, E, 1.f, 1, 1);
    ln_kernel<<<MT, 128, 0, s>>>(cf1, cf2, F(32), F(33), 128, 1);           // LN + GELU
    gemm<true, 0>(s, cf2, 128, 0, 0, F(34), 128, 0, 0, out_cf, 64, 0, 0, F(35), nullptr, MT, 64, 128, 1.f, 1, 1);
    l2norm_kernel<<<MT, 64, 0, s>>>(out_cf, nv, 64);
    gemm<true, 2>(s, nv, 64, (long long)Nn * 64, 0, nv, 64, (long long)Nn * 64, 0,
                  out_sim, Nn, (long long)Nn * Nn, 0, nullptr, nullptr,
                  Nn, Nn, 64, 1.f, Bn, 1);                                  // sim-clip fused

    cond_kernel<<<1, 256, 0, s>>>(keymask, Bn * Nn, flag);

    // ---- aggregate block 1 (top-K most similar) ----
    topk_mask_kernel<<<dim3(Nn, Bn), 128, 0, s>>>(out_sim, mmask, flag, madd, Nn, TOK, 0);
    mha(x, x, F(36), F(37), F(38), F(39), HAG, nullptr, madd, nullptr, y);  // a -> y
    gemm<true, 0>(s, y, E, 0, 0, F(40), E, 0, 0, t, E, 0, 0, F(41), nullptr, MT, E, E, 1.f, 1, 1);
    ln_kernel<<<MT, 128, 0, s>>>(t, y, F(42), F(43), E, 1);                 // LN + GELU
    gemm<true, 0>(s, y, E, 0, 0, F(44), E, 0, 0, x, E, 0, 0, F(45), x, MT, E, E, 1.f, 1, 1);

    // ---- aggregate block 2 (top-K least similar) ----
    topk_mask_kernel<<<dim3(Nn, Bn), 128, 0, s>>>(out_sim, mmask, flag, madd, Nn, TOK, 1);
    mha(x, x, F(46), F(47), F(48), F(49), HAG, nullptr, madd, nullptr, y);
    gemm<true, 0>(s, y, E, 0, 0, F(50), E, 0, 0, t, E, 0, 0, F(51), nullptr, MT, E, E, 1.f, 1, 1);
    ln_kernel<<<MT, 128, 0, s>>>(t, y, F(52), F(53), E, 1);
    gemm<true, 0>(s, y, E, 0, 0, F(54), E, 0, 0, out_feats, E, 0, 0, F(55), x, MT, E, E, 1.f, 1, 1);

    (void)in_sizes; (void)n_in; (void)out_size; (void)ws_size;
}
